// UserGraphConstructor_90305982365986
// MI455X (gfx1250) — matrix operand order repaired and checked by reading, not compiled, not run on hardware
//
#include <hip/hip_runtime.h>
#include <hip/hip_bf16.h>
#include <math.h>

typedef __attribute__((ext_vector_type(16))) __bf16 v16bf;
typedef __attribute__((ext_vector_type(8)))  __bf16 v8bf;
typedef __attribute__((ext_vector_type(4)))  __bf16 v4bf;
typedef __attribute__((ext_vector_type(8)))  float  v8f;

#define S_DIM 256
#define E_DIM 64

// LDS layout (dynamic): items bf16 [256][64] = 32 KB, scores f32 [256][256] = 256 KB,
// tw [256], iw [256], redV [256], redI [256]  -> 299,008 bytes (< 320 KB WGP LDS)
#define SMEM_BYTES (32768 + 262144 + 4 * 1024)

__global__ __launch_bounds__(256) void ugc_fused_kernel(
    const float* __restrict__ itemsG,   // [B,S,E] f32
    const float* __restrict__ tsG,      // [B,S]
    const float* __restrict__ iwG,      // [B,S]
    const float* __restrict__ W1,       // [3,16]
    const float* __restrict__ B1,       // [16]
    const float* __restrict__ W2,       // [16,1]
    const float* __restrict__ B2,       // [1]
    const int*   __restrict__ kptr,     // scalar k
    int*   __restrict__ outSrc,         // [B*k]
    int*   __restrict__ outDst,         // [B*k]
    float* __restrict__ outW)           // [B*k]
{
    extern __shared__ char smem[];
    __bf16* itemsL = (__bf16*)smem;                        // 32768 B
    float*  scores = (float*)(smem + 32768);               // 262144 B
    float*  twL    = (float*)(smem + 32768 + 262144);
    float*  iwL    = twL + S_DIM;
    float*  redV   = iwL + S_DIM;
    int*    redI   = (int*)(redV + S_DIM);

    const int t    = threadIdx.x;
    const int b    = blockIdx.x;
    const int lane = t & 31;
    const int wave = t >> 5;
    const int m    = lane & 15;   // M (for A) / N (for B,C)
    const int hh   = lane >> 4;   // lane half

    // ---- MLP weights into registers (uniform addresses) ----
    float w1r[48], b1r[16], w2r[16];
#pragma unroll
    for (int i = 0; i < 48; ++i) w1r[i] = W1[i];
#pragma unroll
    for (int i = 0; i < 16; ++i) b1r[i] = B1[i];
#pragma unroll
    for (int i = 0; i < 16; ++i) w2r[i] = W2[i];
    const float b2r = B2[0];

    // ---- phase 0: stage items (f32 -> bf16), timestamps, interaction weights ----
    {
        const float4* src = (const float4*)(itemsG + (size_t)b * (S_DIM * E_DIM));
#pragma unroll
        for (int it = 0; it < 16; ++it) {
            const int q = it * 256 + t;          // float4 index, coalesced
            const float4 v = src[q];
            v4bf o;
            o.x = (__bf16)v.x; o.y = (__bf16)v.y; o.z = (__bf16)v.z; o.w = (__bf16)v.w;
            *(v4bf*)(itemsL + q * 4) = o;
        }
        redV[t] = tsG[(size_t)b * S_DIM + t];
        iwL[t]  = iwG[(size_t)b * S_DIM + t];
    }
    __syncthreads();
    // max timestamp reduction
    for (int s2 = 128; s2 > 0; s2 >>= 1) {
        if (t < s2) redV[t] = fmaxf(redV[t], redV[t + s2]);
        __syncthreads();
    }
    const float maxT = redV[0];
    {
        const float ts = tsG[(size_t)b * S_DIM + t];
        // 0.95^(maxT - ts) = exp(ln(0.95)*(maxT-ts))
        twL[t] = __expf(-0.05129329438755058f * (maxT - ts));
    }
    __syncthreads();

    // ---- phase 1: sims = items . items^T via WMMA bf16 -> f32 (K=64 = 2 steps) ----
    // wave w owns row tiles 2w and 2w+1 (rows 32w .. 32w+31)
    for (int rt = wave * 2; rt < wave * 2 + 2; ++rt) {
        const int i0 = rt * 16;
        // A operand (16x32 bf16 per step). Lane(L): M = L%16, half h = L/16.
        // element e : K = kk + 16h + e (contiguous)
        v16bf a0, a1;
        {
            const __bf16* rp = itemsL + (i0 + m) * E_DIM;
            const v8bf lo0 = *(const v8bf*)(rp + 16 * hh);
            const v8bf hi0 = *(const v8bf*)(rp + 16 * hh + 8);
            const v8bf lo1 = *(const v8bf*)(rp + 32 + 16 * hh);
            const v8bf hi1 = *(const v8bf*)(rp + 32 + 16 * hh + 8);
#pragma unroll
            for (int e = 0; e < 8; ++e) {
                a0[e] = lo0[e]; a0[e + 8] = hi0[e];
                a1[e] = lo1[e]; a1[e + 8] = hi1[e];
            }
        }
        for (int j0 = 0; j0 < S_DIM; j0 += 16) {
            // B operand (32x16 bf16): lane gives N = L%16; K = kk + 16h + e (contiguous)
            const v16bf bm0 = *(const v16bf*)(itemsL + (j0 + m) * E_DIM + 16 * hh);
            const v16bf bm1 = *(const v16bf*)(itemsL + (j0 + m) * E_DIM + 32 + 16 * hh);
            v8f c = {};
            c = __builtin_amdgcn_wmma_f32_16x16x32_bf16(false, a0, false, bm0,
                                                        (short)0, c, false, false);
            c = __builtin_amdgcn_wmma_f32_16x16x32_bf16(false, a1, false, bm1,
                                                        (short)0, c, false, false);
            // C/D layout: VGPR r -> row i0 + r + 8h, col j0 + (lane%16)
#pragma unroll
            for (int r = 0; r < 8; ++r)
                scores[(i0 + r + 8 * hh) * S_DIM + j0 + m] = c[r];
        }
    }
    __syncthreads();

    // ---- phases 2+3: per-row L2 norm, then fused MLP + sigmoid + triu/threshold mask ----
    {
        float* row = scores + t * S_DIM;
        float ss = 0.f;
        for (int j = 0; j < S_DIM; j += 4) {
            const float4 v = *(const float4*)(row + j);
            ss = fmaf(v.x, v.x, fmaf(v.y, v.y, fmaf(v.z, v.z, fmaf(v.w, v.w, ss))));
        }
        const float inv = 1.0f / fmaxf(sqrtf(ss), 1e-12f);
        const float twi = twL[t];
        const float iwi = iwL[t];
        for (int j = 0; j < S_DIM; j += 4) {
            const float4 sv = *(const float4*)(row + j);
            const float4 tj = *(const float4*)(twL + j);
            const float sj[4]  = {sv.x, sv.y, sv.z, sv.w};
            const float tw4[4] = {tj.x, tj.y, tj.z, tj.w};
            float o[4];
#pragma unroll
            for (int u = 0; u < 4; ++u) {
                const float f0 = sj[u] * inv;
                const float f1 = twi * tw4[u];
                const float f2 = iwi;
                float acc = b2r;
#pragma unroll
                for (int n = 0; n < 16; ++n) {
                    float hn = fmaf(f0, w1r[n],
                               fmaf(f1, w1r[16 + n],
                               fmaf(f2, w1r[32 + n], b1r[n])));
                    hn  = fmaxf(hn, 0.f);
                    acc = fmaf(hn, w2r[n], acc);
                }
                const float sg = 1.0f / (1.0f + __expf(-acc));
                const int jj = j + u;
                o[u] = (jj > t && sg > 0.1f) ? sg : -__builtin_inff();
            }
            const float4 ov = {o[0], o[1], o[2], o[3]};
            *(float4*)(row + j) = ov;
        }
    }
    __syncthreads();

    // ---- phase 4: iterative top-k over the LDS score matrix ----
    const int k = *kptr;
    for (int q = 0; q < k; ++q) {
        const float* row = scores + t * S_DIM;
        float best = -__builtin_inff();
        int   bi   = t * S_DIM;
        for (int j = 0; j < S_DIM; ++j) {
            const float v = row[j];
            if (v > best) { best = v; bi = t * S_DIM + j; }
        }
        redV[t] = best; redI[t] = bi;
        __syncthreads();
        for (int s2 = 128; s2 > 0; s2 >>= 1) {
            if (t < s2 && redV[t + s2] > redV[t]) {
                redV[t] = redV[t + s2]; redI[t] = redI[t + s2];
            }
            __syncthreads();
        }
        if (t == 0) {
            const float v  = redV[0];
            const int  idx = redI[0];
            const bool valid = v > -__builtin_inff();
            const int pos = b * k + q;
            outSrc[pos] = valid ? (idx >> 8) + b * S_DIM : 0;
            outDst[pos] = valid ? (idx & 255) + b * S_DIM : 0;
            outW[pos]   = valid ? v : 0.f;
            scores[idx] = -__builtin_inff();
        }
        __syncthreads();
    }
}

extern "C" void kernel_launch(void* const* d_in, const int* in_sizes, int n_in,
                              void* d_out, int out_size, void* d_ws, size_t ws_size,
                              hipStream_t stream) {
    (void)d_ws; (void)ws_size; (void)n_in;
    // d_in[0] = user_embeddings (unused by the reference output)
    const float* items = (const float*)d_in[1];
    const float* ts    = (const float*)d_in[2];
    const float* iw    = (const float*)d_in[3];
    const float* W1    = (const float*)d_in[4];
    const float* B1    = (const float*)d_in[5];
    const float* W2    = (const float*)d_in[6];
    const float* B2    = (const float*)d_in[7];
    const int*   kptr  = (const int*)d_in[8];

    const int B  = in_sizes[2] / S_DIM;      // timestamps is [B,S]
    const int BK = out_size / 3;             // out = src[B*k] | dst[B*k] | w[B*k]
    int*   outSrc = (int*)d_out;
    int*   outDst = outSrc + BK;
    float* outW   = (float*)d_out + 2 * BK;

    // Allow ~292 KB dynamic LDS (WGP supports up to 320 KB per workgroup).
    (void)hipFuncSetAttribute((const void*)ugc_fused_kernel,
                              hipFuncAttributeMaxDynamicSharedMemorySize,
                              (int)SMEM_BYTES);

    ugc_fused_kernel<<<B, 256, SMEM_BYTES, stream>>>(
        items, ts, iw, W1, B1, W2, B2, kptr, outSrc, outDst, outW);
}
